// EncoderDecoder_11544872092028
// MI455X (gfx1250) — matrix operand order repaired and checked by reading, not compiled, not run on hardware
//
#include <hip/hip_runtime.h>
#include <hip/hip_bf16.h>
#include <stdint.h>
#include <math.h>

// ---------------- CDNA5 WMMA types ----------------
typedef __bf16 bf16_t;
typedef __attribute__((ext_vector_type(16))) __bf16 v16bf;
typedef __attribute__((ext_vector_type(8)))  __bf16 v8bf;
typedef __attribute__((ext_vector_type(8)))  float  v8f;

#define HID   512
#define GATES 2048
#define BATCH 64
#define SRCL  256
#define VOCAB 32000
#define TMAX  32

// ---- A-tile loader: 16x32 bf16, row-major source with leading dim `ld`.
// ISA layout (CDNA5 ISA, 16-bit A 16x32):
//   lanes 0-15 : row M=lane,   K = 0..7 and 16..23
//   lanes 16-31: row M=lane-16, K = 8..15 and 24..31
__device__ __forceinline__ v16bf load_a16(const bf16_t* __restrict__ base, int ld) {
    const int lane = threadIdx.x & 31;
    const int row  = lane & 15;
    const int kh   = (lane >> 4) << 3;        // 0 or 8
    const bf16_t* p = base + row * ld + kh;
    v8bf lo = *reinterpret_cast<const v8bf*>(p);
    v8bf hi = *reinterpret_cast<const v8bf*>(p + 16);
    v16bf a;
#pragma unroll
    for (int i = 0; i < 8; ++i) { a[i] = lo[i]; a[i + 8] = hi[i]; }
    return a;
}

// ---- B-tile loader: B[k][n] = WT[n][k] (weights stored output-feature-major).
// Lane<16: N=lane, K=0..7 and 16..23; lane>=16: N=lane-16, K=8..15 and 24..31.
__device__ __forceinline__ v16bf load_b16(const bf16_t* __restrict__ baseT, int ld) {
    const int lane = threadIdx.x & 31;
    const int n  = lane & 15;
    const int ko = (lane >> 4) << 3;          // 0 or 8
    const bf16_t* p = baseT + n * ld + ko;
    v8bf lo = *reinterpret_cast<const v8bf*>(p);
    v8bf hi = *reinterpret_cast<const v8bf*>(p + 16);
    v16bf b;
#pragma unroll
    for (int i = 0; i < 8; ++i) { b[i] = lo[i]; b[i + 8] = hi[i]; }
    return b;
}

__device__ __forceinline__ v8f wmma_bf16(v16bf a, v16bf b, v8f c) {
    return __builtin_amdgcn_wmma_f32_16x16x32_bf16(false, a, false, b, (short)0, c,
                                                   false, false);
}

__device__ __forceinline__ float sigmoidf_(float x) { return 1.0f / (1.0f + __expf(-x)); }

// Async copy of 16 bytes/lane from global to LDS (GLOBAL_LOAD_ASYNC_TO_LDS_B128,
// tracked by ASYNCcnt). lds_off = byte offset inside the workgroup LDS allocation.
__device__ __forceinline__ void async_g2l_b128(unsigned lds_off, const void* gaddr) {
    asm volatile("global_load_async_to_lds_b128 %0, %1, off"
                 :: "v"(lds_off), "v"((unsigned long long)(uintptr_t)gaddr)
                 : "memory");
}
__device__ __forceinline__ void wait_asynccnt0() {
    asm volatile("s_wait_asynccnt 0x0" ::: "memory");
}

// =====================================================================
// Fused encoder: one launch for all 256 timesteps.
// grid = 4 blocks (one 16-row batch tile each) x 1024 threads (32 waves).
// Wave w owns hidden cols [16w,16w+16); its c/h f32 fragments persist in
// registers; h(bf16) is shared through double-buffered LDS; x_t tiles are
// prefetched one step ahead with async global->LDS loads.
// =====================================================================
__global__ __launch_bounds__(1024)
void encoder_fused_kernel(const bf16_t* __restrict__ Xb,     // (L*B,512) bf16 embeds
                          const int*    __restrict__ source, // (L,B)
                          const bf16_t* __restrict__ Wih,    // (2048,512)
                          const bf16_t* __restrict__ Whh,    // (2048,512)
                          const float*  __restrict__ bias,   // (2048,)
                          bf16_t* __restrict__ hsb,          // (L*B,512) masked outputs
                          float*  __restrict__ h_out,        // (64,512)
                          float*  __restrict__ c_out,        // (64,512)
                          bf16_t* __restrict__ hb_out)       // (64,512)
{
    __shared__ bf16_t xsh[2][16 * HID];   // 2 x 16KB
    __shared__ bf16_t hsh[2][16 * HID];   // 2 x 16KB

    const int mt   = blockIdx.x;          // batch tile 0..3
    const int tid  = threadIdx.x;
    const int wv   = tid >> 5;            // wave id == hidden n-tile 0..31
    const int lane = tid & 31;
    const int nb   = wv * 16;
    const int n    = lane & 15;
    const int hi2  = lane >> 4;
    const int j    = nb + n;

    float creg[8], hreg[8];
#pragma unroll
    for (int r = 0; r < 8; ++r) { creg[r] = 0.f; hreg[r] = 0.f; }

    // zero h buffer 0
    for (int i = tid; i < 16 * HID; i += 1024) hsh[0][i] = (bf16_t)0.f;

    // async-load x tile for t=0 (each thread moves 16B; 1024*16B = 16KB tile)
    {
        const bf16_t* g = Xb + ((size_t)mt * 16) * HID + tid * 8;
        async_g2l_b128((unsigned)(uintptr_t)&xsh[0][tid * 8], g);
        wait_asynccnt0();
    }
    __syncthreads();

    const float bi  = bias[j];
    const float bff = bias[HID + j];
    const float bg  = bias[2 * HID + j];
    const float bo  = bias[3 * HID + j];

#pragma unroll 1
    for (int t = 0; t < SRCL; ++t) {
        const int cur = t & 1, nxt = cur ^ 1;

        // prefetch x[t+1] into the other LDS buffer (overlaps with this step's WMMAs)
        if (t + 1 < SRCL) {
            const bf16_t* g = Xb + ((size_t)(t + 1) * BATCH + mt * 16) * HID + tid * 8;
            async_g2l_b128((unsigned)(uintptr_t)&xsh[nxt][tid * 8], g);
        }

        v8f a0 = {}, a1 = {}, a2 = {}, a3 = {};
#pragma unroll 4
        for (int k0 = 0; k0 < HID; k0 += 32) {   // x contribution (A from LDS)
            v16bf av = load_a16(&xsh[cur][k0], HID);
            a0 = wmma_bf16(av, load_b16(Wih + (nb          ) * HID + k0, HID), a0);
            a1 = wmma_bf16(av, load_b16(Wih + (nb +     HID) * HID + k0, HID), a1);
            a2 = wmma_bf16(av, load_b16(Wih + (nb + 2 * HID) * HID + k0, HID), a2);
            a3 = wmma_bf16(av, load_b16(Wih + (nb + 3 * HID) * HID + k0, HID), a3);
        }
#pragma unroll 4
        for (int k0 = 0; k0 < HID; k0 += 32) {   // h contribution (A from LDS)
            v16bf av = load_a16(&hsh[cur][k0], HID);
            a0 = wmma_bf16(av, load_b16(Whh + (nb          ) * HID + k0, HID), a0);
            a1 = wmma_bf16(av, load_b16(Whh + (nb +     HID) * HID + k0, HID), a1);
            a2 = wmma_bf16(av, load_b16(Whh + (nb + 2 * HID) * HID + k0, HID), a2);
            a3 = wmma_bf16(av, load_b16(Whh + (nb + 3 * HID) * HID + k0, HID), a3);
        }

        // LSTM cell + pack/pad masking; C layout: lane,VGPR r -> M=r+8*hi2, N=lane&15
#pragma unroll
        for (int r = 0; r < 8; ++r) {
            const int m = r + 8 * hi2;
            const int b = mt * 16 + m;
            const float iv = sigmoidf_(a0[r] + bi);
            const float fv = sigmoidf_(a1[r] + bff);
            const float gv = tanhf(a2[r] + bg);
            const float ov = sigmoidf_(a3[r] + bo);
            const float cold = creg[r];
            const float cn = fv * cold + iv * gv;
            const float hn = ov * tanhf(cn);
            const float mk = (source[t * BATCH + b] > 0) ? 1.0f : 0.0f;
            creg[r] = mk * cn + (1.0f - mk) * cold;
            const float hw = mk * hn + (1.0f - mk) * hreg[r];
            hreg[r] = hw;
            hsh[nxt][m * HID + j] = (bf16_t)hw;
            hsb[((size_t)t * BATCH + b) * HID + j] = (bf16_t)(hn * mk);
        }

        wait_asynccnt0();     // x[t+1] landed
        __syncthreads();      // h[t+1] + x[t+1] visible to all waves
    }

    // publish final state for the decoder
#pragma unroll
    for (int r = 0; r < 8; ++r) {
        const int b = mt * 16 + r + 8 * hi2;
        h_out[b * HID + j]  = hreg[r];
        c_out[b * HID + j]  = creg[r];
        hb_out[b * HID + j] = (bf16_t)hreg[r];
    }
}

// ---------------- Decoder LSTM step (multi-WG, WMMA) ----------------
__global__ __launch_bounds__(32)
void lstm_step_kernel(const bf16_t* __restrict__ xb,
                      const bf16_t* __restrict__ hb_in,
                      bf16_t* __restrict__ hb_out,
                      float* __restrict__ h,
                      float* __restrict__ c,
                      const bf16_t* __restrict__ Wih,
                      const bf16_t* __restrict__ Whh,
                      const float* __restrict__ bias)
{
    const int mt = blockIdx.x;
    const int nt = blockIdx.y;
    const int nb = nt * 16;

    v8f a0 = {}, a1 = {}, a2 = {}, a3 = {};
#pragma unroll 4
    for (int k0 = 0; k0 < 2 * HID; k0 += 32) {
        const bf16_t* Ab = (k0 < HID) ? xb : hb_in;
        const int kk = k0 & (HID - 1);
        v16bf av = load_a16(Ab + mt * 16 * HID + kk, HID);
        const bf16_t* W = (k0 < HID) ? Wih : Whh;
        a0 = wmma_bf16(av, load_b16(W + (nb          ) * HID + kk, HID), a0);
        a1 = wmma_bf16(av, load_b16(W + (nb +     HID) * HID + kk, HID), a1);
        a2 = wmma_bf16(av, load_b16(W + (nb + 2 * HID) * HID + kk, HID), a2);
        a3 = wmma_bf16(av, load_b16(W + (nb + 3 * HID) * HID + kk, HID), a3);
    }

    const int lane = threadIdx.x & 31;
    const int n = lane & 15, hi2 = lane >> 4;
    const int j = nb + n;
    const float bi  = bias[j];
    const float bff = bias[HID + j];
    const float bg  = bias[2 * HID + j];
    const float bo  = bias[3 * HID + j];

#pragma unroll
    for (int r = 0; r < 8; ++r) {
        const int b = mt * 16 + r + 8 * hi2;
        const int idx = b * HID + j;
        const float iv = sigmoidf_(a0[r] + bi);
        const float fv = sigmoidf_(a1[r] + bff);
        const float gv = tanhf(a2[r] + bg);
        const float ov = sigmoidf_(a3[r] + bo);
        const float cn = fv * c[idx] + iv * gv;
        const float hn = ov * tanhf(cn);
        c[idx] = cn;
        h[idx] = hn;
        hb_out[idx] = (bf16_t)hn;
    }
}

// ---------------- hsW = hs @ W_a  (16384 x 512 x 512), WMMA ----------------
__global__ __launch_bounds__(32)
void hsw_kernel(const bf16_t* __restrict__ hsb,
                const bf16_t* __restrict__ WaT,
                float* __restrict__ hsW)
{
    const int mt = blockIdx.x;   // 0..1023
    const int nt = blockIdx.y;   // 0..31
    v8f acc = {};
#pragma unroll 4
    for (int k0 = 0; k0 < HID; k0 += 32) {
        v16bf a = load_a16(hsb + (size_t)mt * 16 * HID + k0, HID);
        v16bf b = load_b16(WaT + (nt * 16) * HID + k0, HID);
        acc = wmma_bf16(a, b, acc);
    }
    const int lane = threadIdx.x & 31;
    const int n = lane & 15, hi2 = lane >> 4;
#pragma unroll
    for (int r = 0; r < 8; ++r) {
        const size_t m = (size_t)mt * 16 + r + 8 * hi2;
        hsW[m * HID + nt * 16 + n] = acc[r];
    }
}

// ---------------- Attention scores + softmax ----------------
__global__ __launch_bounds__(256)
void attn_scores_kernel(const float* __restrict__ hd,
                        const float* __restrict__ hsW,
                        const int* __restrict__ source,
                        float* __restrict__ attw)
{
    const int b = blockIdx.x;
    const int l = threadIdx.x;     // 256 = SRCL
    __shared__ float sh[HID];
    __shared__ float red[256];
    for (int k = l; k < HID; k += 256) sh[k] = hd[b * HID + k];
    __syncthreads();
    const float* w = hsW + ((size_t)l * BATCH + b) * HID;
    float s = 0.f;
#pragma unroll 8
    for (int k = 0; k < HID; ++k) s += sh[k] * w[k];
    red[l] = s;
    __syncthreads();
    for (int st = 128; st > 0; st >>= 1) {
        if (l < st) red[l] = fmaxf(red[l], red[l + st]);
        __syncthreads();
    }
    const float mx = red[0];
    __syncthreads();
    const float mk = (source[l * BATCH + b] > 0) ? 1.0f : 0.0f;
    const float e = __expf(s - mx) * mk;
    red[l] = e;
    __syncthreads();
    for (int st = 128; st > 0; st >>= 1) {
        if (l < st) red[l] += red[l + st];
        __syncthreads();
    }
    attw[b * SRCL + l] = e / red[0];
}

// ---------------- ctx[b][j] = sum_l a[b][l]*hs[l][b][j] ----------------
__global__ __launch_bounds__(512)
void attn_ctx_kernel(const float* __restrict__ attw,
                     const bf16_t* __restrict__ hsb,
                     bf16_t* __restrict__ ctxb)
{
    const int b = blockIdx.x;
    const int j = threadIdx.x;   // 512
    __shared__ float av[SRCL];
    if (j < SRCL) av[j] = attw[b * SRCL + j];
    __syncthreads();
    float s = 0.f;
#pragma unroll 4
    for (int l = 0; l < SRCL; ++l)
        s += av[l] * (float)hsb[((size_t)l * BATCH + b) * HID + j];
    ctxb[b * HID + j] = (bf16_t)s;
}

// ---------------- att = tanh([ctx|hd] @ W_c + b_c), WMMA K=1024 ----------------
__global__ __launch_bounds__(32)
void attn_combine_kernel(const bf16_t* __restrict__ ctxb,
                         const bf16_t* __restrict__ hdb,
                         const bf16_t* __restrict__ WcT,
                         const float* __restrict__ bc,
                         bf16_t* __restrict__ attb)
{
    const int mt = blockIdx.x, nt = blockIdx.y;
    v8f acc = {};
#pragma unroll 4
    for (int k0 = 0; k0 < 2 * HID; k0 += 32) {
        const bf16_t* Ab = (k0 < HID) ? ctxb : hdb;
        const int kk = k0 & (HID - 1);
        v16bf a = load_a16(Ab + mt * 16 * HID + kk, HID);
        v16bf b = load_b16(WcT + (nt * 16) * (2 * HID) + k0, 2 * HID);
        acc = wmma_bf16(a, b, acc);
    }
    const int lane = threadIdx.x & 31;
    const int n = lane & 15, hi2 = lane >> 4;
    const int j = nt * 16 + n;
    const float bj = bc[j];
#pragma unroll
    for (int r = 0; r < 8; ++r) {
        const int b = mt * 16 + r + 8 * hi2;
        attb[b * HID + j] = (bf16_t)tanhf(acc[r] + bj);
    }
}

// ---------------- logits = att @ out_W^T + out_b (64 x 32000 x 512) ----------------
// 4 N-tiles per wave -> 4x A-fragment reuse per k-iteration. grid (4, 500).
__global__ __launch_bounds__(32)
void logits_kernel(const bf16_t* __restrict__ attb,
                   const bf16_t* __restrict__ oWb,
                   const float* __restrict__ ob,
                   float* __restrict__ out)
{
    const int mt  = blockIdx.x;          // 0..3
    const int nt4 = blockIdx.y * 4;      // 0..1996
    const bf16_t* W0 = oWb + (size_t)(nt4 + 0) * 16 * HID;
    const bf16_t* W1 = oWb + (size_t)(nt4 + 1) * 16 * HID;
    const bf16_t* W2 = oWb + (size_t)(nt4 + 2) * 16 * HID;
    const bf16_t* W3 = oWb + (size_t)(nt4 + 3) * 16 * HID;
    v8f a0 = {}, a1 = {}, a2 = {}, a3 = {};
#pragma unroll 4
    for (int k0 = 0; k0 < HID; k0 += 32) {
        v16bf av = load_a16(attb + mt * 16 * HID + k0, HID);
        a0 = wmma_bf16(av, load_b16(W0 + k0, HID), a0);
        a1 = wmma_bf16(av, load_b16(W1 + k0, HID), a1);
        a2 = wmma_bf16(av, load_b16(W2 + k0, HID), a2);
        a3 = wmma_bf16(av, load_b16(W3 + k0, HID), a3);
    }
    const int lane = threadIdx.x & 31;
    const int n = lane & 15, hi2 = lane >> 4;
#pragma unroll
    for (int q = 0; q < 4; ++q) {
        const v8f& acc = (q == 0) ? a0 : (q == 1) ? a1 : (q == 2) ? a2 : a3;
        const int col = (nt4 + q) * 16 + n;
        const float bj = ob[col];
#pragma unroll
        for (int r = 0; r < 8; ++r) {
            const int b = mt * 16 + r + 8 * hi2;
            out[(size_t)b * VOCAB + col] = acc[r] + bj;
        }
    }
}

// ---------------- argmax over vocab ----------------
__global__ __launch_bounds__(256)
void argmax_kernel(const float* __restrict__ logits, int* __restrict__ y)
{
    const int b = blockIdx.x;
    const float* row = logits + (size_t)b * VOCAB;
    __shared__ float sv[256];
    __shared__ int si[256];
    float best = -__builtin_inff();
    int bidx = 0x7fffffff;
    for (int i = threadIdx.x; i < VOCAB; i += 256) {
        const float v = row[i];
        if (v > best || (v == best && i < bidx)) { best = v; bidx = i; }
    }
    sv[threadIdx.x] = best; si[threadIdx.x] = bidx;
    __syncthreads();
    for (int st = 128; st > 0; st >>= 1) {
        if (threadIdx.x < st) {
            const float v2 = sv[threadIdx.x + st];
            const int i2 = si[threadIdx.x + st];
            if (v2 > sv[threadIdx.x] || (v2 == sv[threadIdx.x] && i2 < si[threadIdx.x])) {
                sv[threadIdx.x] = v2; si[threadIdx.x] = i2;
            }
        }
        __syncthreads();
    }
    if (threadIdx.x == 0) y[b] = si[0];
}

// ---------------- setup / conversion kernels ----------------
__global__ void f32_to_bf16_kernel(const float* __restrict__ s, bf16_t* __restrict__ d, int n) {
    const int i = blockIdx.x * blockDim.x + threadIdx.x;
    if (i < n) d[i] = (bf16_t)s[i];
}
__global__ void transpose_bf16_kernel(const float* __restrict__ s, bf16_t* __restrict__ d,
                                      int R, int C) {  // s: RxC -> d: CxR
    const int i = blockIdx.x * blockDim.x + threadIdx.x;
    if (i < R * C) { const int r = i / C, cc = i % C; d[cc * R + r] = (bf16_t)s[i]; }
}
__global__ void bias_sum_kernel(const float* __restrict__ a, const float* __restrict__ b,
                                float* __restrict__ o, int n) {
    const int i = blockIdx.x * blockDim.x + threadIdx.x;
    if (i < n) o[i] = a[i] + b[i];
}
__global__ void gather_embed_kernel(const int* __restrict__ tok, const float* __restrict__ emb,
                                    bf16_t* __restrict__ out, int ntok) {
    const int idx = blockIdx.x * blockDim.x + threadIdx.x;
    if (idx < ntok * HID) {
        const int i = idx >> 9, k = idx & (HID - 1);
        out[idx] = (bf16_t)emb[(size_t)tok[i] * HID + k];
    }
}
__global__ void set_bos_kernel(int* __restrict__ y) {
    if (threadIdx.x < BATCH) y[threadIdx.x] = 1;
}

// ---------------- host orchestration ----------------
static inline size_t align256(size_t x) { return (x + 255) & ~(size_t)255; }

extern "C" void kernel_launch(void* const* d_in, const int* in_sizes, int n_in,
                              void* d_out, int out_size, void* d_ws, size_t ws_size,
                              hipStream_t stream) {
    (void)in_sizes; (void)n_in; (void)out_size; (void)ws_size;

    const int*   source  = (const int*)  d_in[0];
    const float* enc_emb = (const float*)d_in[1];
    const float* enc_Wih = (const float*)d_in[2];
    const float* enc_Whh = (const float*)d_in[3];
    const float* enc_bih = (const float*)d_in[4];
    const float* enc_bhh = (const float*)d_in[5];
    const float* dec_emb = (const float*)d_in[6];
    const float* dec_Wih = (const float*)d_in[7];
    const float* dec_Whh = (const float*)d_in[8];
    const float* dec_bih = (const float*)d_in[9];
    const float* dec_bhh = (const float*)d_in[10];
    const float* W_a     = (const float*)d_in[11];
    const float* W_c     = (const float*)d_in[12];
    const float* b_c     = (const float*)d_in[13];
    const float* out_W   = (const float*)d_in[14];
    const float* out_b   = (const float*)d_in[15];
    float* out = (float*)d_out;

    // ---- workspace arena (~112 MB) ----
    uint8_t* ws = (uint8_t*)d_ws;
    size_t off = 0;
    auto alloc = [&](size_t bytes) -> void* {
        void* p = ws + off;
        off = align256(off + bytes);
        return p;
    };
    bf16_t* encWih_b = (bf16_t*)alloc((size_t)GATES * HID * 2);
    bf16_t* encWhh_b = (bf16_t*)alloc((size_t)GATES * HID * 2);
    bf16_t* decWih_b = (bf16_t*)alloc((size_t)GATES * HID * 2);
    bf16_t* decWhh_b = (bf16_t*)alloc((size_t)GATES * HID * 2);
    bf16_t* WaT_b    = (bf16_t*)alloc((size_t)HID * HID * 2);
    bf16_t* WcT_b    = (bf16_t*)alloc((size_t)HID * 2 * HID * 2);
    bf16_t* oW_b     = (bf16_t*)alloc((size_t)VOCAB * HID * 2);
    float*  bias_e   = (float*) alloc((size_t)GATES * 4);
    float*  bias_d   = (float*) alloc((size_t)GATES * 4);
    bf16_t* Xb       = (bf16_t*)alloc((size_t)SRCL * BATCH * HID * 2);
    bf16_t* hsb      = (bf16_t*)alloc((size_t)SRCL * BATCH * HID * 2);
    float*  hsW      = (float*) alloc((size_t)SRCL * BATCH * HID * 4);
    float*  h        = (float*) alloc((size_t)BATCH * HID * 4);
    float*  c        = (float*) alloc((size_t)BATCH * HID * 4);
    bf16_t* hb0      = (bf16_t*)alloc((size_t)BATCH * HID * 2);
    bf16_t* hb1      = (bf16_t*)alloc((size_t)BATCH * HID * 2);
    bf16_t* xdec     = (bf16_t*)alloc((size_t)BATCH * HID * 2);
    float*  attw     = (float*) alloc((size_t)BATCH * SRCL * 4);
    bf16_t* ctxb     = (bf16_t*)alloc((size_t)BATCH * HID * 2);
    bf16_t* attb     = (bf16_t*)alloc((size_t)BATCH * HID * 2);
    int*    y        = (int*)   alloc((size_t)BATCH * 4);

    auto nb = [](int n) { return (n + 255) / 256; };

    // ---- weight conversion ----
    f32_to_bf16_kernel<<<nb(GATES * HID), 256, 0, stream>>>(enc_Wih, encWih_b, GATES * HID);
    f32_to_bf16_kernel<<<nb(GATES * HID), 256, 0, stream>>>(enc_Whh, encWhh_b, GATES * HID);
    f32_to_bf16_kernel<<<nb(GATES * HID), 256, 0, stream>>>(dec_Wih, decWih_b, GATES * HID);
    f32_to_bf16_kernel<<<nb(GATES * HID), 256, 0, stream>>>(dec_Whh, decWhh_b, GATES * HID);
    f32_to_bf16_kernel<<<nb(VOCAB * HID), 256, 0, stream>>>(out_W, oW_b, VOCAB * HID);
    transpose_bf16_kernel<<<nb(HID * HID), 256, 0, stream>>>(W_a, WaT_b, HID, HID);
    transpose_bf16_kernel<<<nb(2 * HID * HID), 256, 0, stream>>>(W_c, WcT_b, 2 * HID, HID);
    bias_sum_kernel<<<nb(GATES), 256, 0, stream>>>(enc_bih, enc_bhh, bias_e, GATES);
    bias_sum_kernel<<<nb(GATES), 256, 0, stream>>>(dec_bih, dec_bhh, bias_d, GATES);

    // ---- encoder: gather embeddings, then one fused kernel for all 256 steps ----
    gather_embed_kernel<<<nb(SRCL * BATCH * HID), 256, 0, stream>>>(source, enc_emb, Xb,
                                                                    SRCL * BATCH);
    encoder_fused_kernel<<<4, 1024, 0, stream>>>(Xb, source, encWih_b, encWhh_b, bias_e,
                                                 hsb, h, c, hb0);

    // hsW = hs @ W_a
    hsw_kernel<<<dim3(1024, 32), 32, 0, stream>>>(hsb, WaT_b, hsW);

    // ---- decoder ----
    set_bos_kernel<<<1, 64, 0, stream>>>(y);
    bf16_t* hb_cur = hb0;
    bf16_t* hb_alt = hb1;
    for (int t = 0; t < TMAX; ++t) {
        gather_embed_kernel<<<nb(BATCH * HID), 256, 0, stream>>>(y, dec_emb, xdec, BATCH);
        lstm_step_kernel<<<dim3(4, 32), 32, 0, stream>>>(
            xdec, hb_cur, hb_alt, h, c, decWih_b, decWhh_b, bias_d);
        bf16_t* tmp = hb_cur; hb_cur = hb_alt; hb_alt = tmp;

        attn_scores_kernel<<<BATCH, 256, 0, stream>>>(h, hsW, source, attw);
        attn_ctx_kernel<<<BATCH, 512, 0, stream>>>(attw, hsb, ctxb);
        attn_combine_kernel<<<dim3(4, 32), 32, 0, stream>>>(ctxb, hb_cur, WcT_b, b_c, attb);

        float* out_t = out + (size_t)t * BATCH * VOCAB;
        logits_kernel<<<dim3(4, 500), 32, 0, stream>>>(attb, oW_b, out_b, out_t);
        argmax_kernel<<<BATCH, 256, 0, stream>>>(out_t, y);
    }
}
